// EncoderClassifier_33157147525457
// MI455X (gfx1250) — matrix operand order repaired and checked by reading, not compiled, not run on hardware
//
#include <hip/hip_runtime.h>
#include <hip/hip_bf16.h>
#include <hip/hip_fp16.h>

// ---------------------------------------------------------------------------
// MI455X (gfx1250, wave32) implementation.
// GEMMs run through v_wmma_f32_16x16x32_f16 (f16 in, fp32 accumulate).
// Row-major operand tiles are staged with GLOBAL_LOAD_ASYNC_TO_LDS_B128
// (ASYNCcnt) via inline asm; transposed tiles use load + ds scatter.
// Residual stream / LN / softmax / BN stay fp32.
// ---------------------------------------------------------------------------

typedef _Float16 v16h __attribute__((ext_vector_type(16)));
typedef _Float16 v8h  __attribute__((ext_vector_type(8)));
typedef float    v8f  __attribute__((ext_vector_type(8)));

#define N_EMBD   512
#define TT       128
#define BATCH    256
#define NHEAD    8
#define HEADD    64
#define NLAYERS  8
#define ROWS     (BATCH * TT)     // 32768
#define FF       (4 * N_EMBD)     // 2048
#define EPSV     1e-5f

__device__ inline v16h mk16(v8h lo, v8h hi) {
    v16h r;
#pragma unroll
    for (int i = 0; i < 8; ++i) { r[i] = lo[i]; r[8 + i] = hi[i]; }
    return r;
}

// Async global->LDS copy of 16 bytes (8 halves). lds_off is the 32-bit LDS
// byte offset (generic LDS pointer truncated per the flat-aperture rule).
__device__ inline void async_b128(unsigned lds_off, const void* gaddr) {
    asm volatile("global_load_async_to_lds_b128 %0, %1, off"
                 :: "v"(lds_off), "v"(gaddr) : "memory");
}
__device__ inline void wait_async0() {
    asm volatile("s_wait_asynccnt 0x0" ::: "memory");
}

// ---------------------------------------------------------------------------
// Embedding: x = tok_emb[index] + pos_emb ; writes fp32 stream + f16 operand
// ---------------------------------------------------------------------------
__global__ __launch_bounds__(256) void k_embed(const int* __restrict__ idx,
                                               const float* __restrict__ tok,
                                               const float* __restrict__ pos,
                                               float* __restrict__ X,
                                               _Float16* __restrict__ Xh) {
    int i   = blockIdx.x * 256 + threadIdx.x;     // covers ROWS*N_EMBD exactly
    int row = i >> 9;
    int c   = i & 511;
    int t   = row & (TT - 1);
    int tk  = idx[row];
    float v = tok[tk * N_EMBD + c] + pos[t * N_EMBD + c];
    X[i]  = v;
    Xh[i] = (_Float16)v;
}

// ---------------------------------------------------------------------------
// fp32 -> f16 cast (weights)
// ---------------------------------------------------------------------------
__global__ __launch_bounds__(256) void k_cast_f16(const float* __restrict__ in,
                                                  _Float16* __restrict__ out, int n) {
    int i = blockIdx.x * 256 + threadIdx.x;
    if (i < n) out[i] = (_Float16)in[i];
}

// Pack Wq/Wk/Wv (L,H,C,K) -> per-layer row-major (C, H*64+K) f16
__global__ __launch_bounds__(256) void k_pack_qkv(const float* __restrict__ in,
                                                  _Float16* __restrict__ out) {
    int i = blockIdx.x * 256 + threadIdx.x;      // 8*512*512 total, exact
    int l = i >> 18;
    int r = i & 262143;
    int c = r >> 9;
    int n = r & 511;
    int h = n >> 6, k = n & 63;
    out[i] = (_Float16)in[((size_t)(l * NHEAD + h) * N_EMBD + c) * HEADD + k];
}

// ---------------------------------------------------------------------------
// WMMA GEMM: C(MxN) = A(MxK) * B(KxN) [+bias] [+relu], A/B f16, acc fp32.
// Workgroup tile 128x128x32, 8 waves (2x4), each wave 64x32 (8 16x16 accs).
// A tile staged async (global->LDS DMA path), B tile load+transpose-scatter.
// ---------------------------------------------------------------------------
#define BM 128
#define BN 128
#define BK 32
#define LDA_S 40   // halves (80B rows -> distinct banks for 16 lanes)
#define LDB_S 40

__global__ __launch_bounds__(256) void wmma_gemm(
        const _Float16* __restrict__ A, const _Float16* __restrict__ Bm,
        const float* __restrict__ bias,
        float* __restrict__ Cf, _Float16* __restrict__ Ch,
        int M, int N, int K, int relu) {
    (void)M;
    __shared__ _Float16 As[BM * LDA_S];
    __shared__ _Float16 Bs[BN * LDB_S];   // stored transposed: Bs[n][k]

    const int tid  = threadIdx.x;
    const int lane = tid & 31;
    const int wave = tid >> 5;
    const int wm   = wave >> 2;           // 0..1
    const int wn   = wave & 3;            // 0..3
    const int l15  = lane & 15;
    const int hi   = lane >> 4;
    const int bm0  = blockIdx.y * BM;
    const int bn0  = blockIdx.x * BN;

    v8f z = {0.f, 0.f, 0.f, 0.f, 0.f, 0.f, 0.f, 0.f};
    v8f acc[4][2];
#pragma unroll
    for (int i = 0; i < 4; ++i)
#pragma unroll
        for (int j = 0; j < 2; ++j) acc[i][j] = z;

    // per-thread staging coordinates
    const int ar   = tid >> 1;            // A tile row 0..127
    const int aseg = (tid & 1) * 16;      // A halves segment (0/16)
    const unsigned alds0 = (unsigned)(size_t)&As[ar * LDA_S + aseg];
    const int bkr  = tid >> 3;            // B tile k-row 0..31
    const int bn0t = (tid & 7) * 16;      // B cols segment

    for (int kk = 0; kk < K; kk += BK) {
        // ---- stage A tile (128x32, row-major) via async global->LDS ----
        {
            const _Float16* gp = A + (size_t)(bm0 + ar) * K + kk + aseg;
            async_b128(alds0, gp);
            async_b128(alds0 + 16u, gp + 8);
            if (kk + BK < K) __builtin_prefetch(gp + BK, 0, 1);
        }
        // ---- stage B tile (32x128) transposed into Bs[n][k] ----
        {
            const _Float16* gp = Bm + (size_t)(kk + bkr) * N + bn0 + bn0t;
            int4 b0 = *(const int4*)(gp);
            int4 b1 = *(const int4*)(gp + 8);
            const _Float16* h0 = (const _Float16*)&b0;
            const _Float16* h1 = (const _Float16*)&b1;
#pragma unroll
            for (int j = 0; j < 8; ++j) Bs[(bn0t + j) * LDB_S + bkr]     = h0[j];
#pragma unroll
            for (int j = 0; j < 8; ++j) Bs[(bn0t + 8 + j) * LDB_S + bkr] = h1[j];
        }
        wait_async0();
        __syncthreads();

        // B fragments for this wave's two 16-col tiles
        v16h bfrag[2];
#pragma unroll
        for (int nt = 0; nt < 2; ++nt) {
            int n = wn * 32 + nt * 16 + l15;
            const _Float16* p = &Bs[n * LDB_S + hi * 16];
            bfrag[nt] = mk16(*(const v8h*)p, *(const v8h*)(p + 8));
        }
#pragma unroll
        for (int mt = 0; mt < 4; ++mt) {
            int m   = wm * 64 + mt * 16 + l15;
            int klo = hi * 16;
            const _Float16* p = &As[m * LDA_S + klo];
            v16h a = mk16(*(const v8h*)p, *(const v8h*)(p + 8));
#pragma unroll
            for (int nt = 0; nt < 2; ++nt) {
                acc[mt][nt] = __builtin_amdgcn_wmma_f32_16x16x32_f16(
                    false, a, false, bfrag[nt], (short)0, acc[mt][nt], false, false);
            }
        }
        __syncthreads();
    }

    // ---- epilogue: bias / relu / fp32 and-or f16 stores ----
#pragma unroll
    for (int mt = 0; mt < 4; ++mt) {
#pragma unroll
        for (int nt = 0; nt < 2; ++nt) {
            int col  = bn0 + wn * 32 + nt * 16 + l15;
            float bv = bias ? bias[col] : 0.0f;
#pragma unroll
            for (int v = 0; v < 8; ++v) {
                int row = bm0 + wm * 64 + mt * 16 + hi * 8 + v;
                float val = acc[mt][nt][v] + bv;
                if (relu) val = val > 0.f ? val : 0.f;
                size_t o = (size_t)row * N + col;
                if (Cf) Cf[o] = val;
                if (Ch) Ch[o] = (_Float16)val;
            }
        }
    }
}

// ---------------------------------------------------------------------------
// Fused attention per (b,h): S=QK^T (WMMA) -> softmax (fp32) -> O=PV (WMMA)
// Dynamic LDS layout (119KB of the 320KB WGP budget):
//   [0,36864)      Qs(128x72h) + Ks(128x72h); reused as P(128x136h) after S
//   [36864,54272)  Vt(64x136h)  (V transposed: Vt[d][t])
//   [54272,121856) S (128x132 f32)
// Q/K tiles staged via async global->LDS; V transposed via ds scatter.
// ---------------------------------------------------------------------------
#define QS  72
#define VTS 136
#define PS  136
#define SS  132
#define ATT_LDS 121856

__global__ __launch_bounds__(256) void k_attention(
        const _Float16* __restrict__ Qh, const _Float16* __restrict__ Kh,
        const _Float16* __restrict__ Vh, _Float16* __restrict__ Oh) {
    extern __shared__ char smem[];
    _Float16* Qs = (_Float16*)smem;
    _Float16* Ks = (_Float16*)(smem + 18432);
    _Float16* Pb = (_Float16*)smem;                   // aliases Qs/Ks (phase 2)
    _Float16* Vt = (_Float16*)(smem + 36864);
    float*    Sb = (float*)(smem + 36864 + 17408);

    const int tid  = threadIdx.x;
    const int lane = tid & 31, wave = tid >> 5;
    const int l15  = lane & 15, hi = lane >> 4;
    const int bh   = blockIdx.x;
    const int b    = bh >> 3, h = bh & 7;
    const size_t rowbase = (size_t)b * TT * N_EMBD + (size_t)h * HEADD;

    // stage Q, K row-major (async DMA) and V transposed (scatter)
#pragma unroll
    for (int it = 0; it < 4; ++it) {
        int idx = it * 256 + tid;
        int t   = idx >> 3;
        int seg = (idx & 7) * 8;
        const size_t g = rowbase + (size_t)t * N_EMBD + seg;
        async_b128((unsigned)(size_t)&Qs[t * QS + seg], Qh + g);
        async_b128((unsigned)(size_t)&Ks[t * QS + seg], Kh + g);
        v8h vv = *(const v8h*)(Vh + g);
#pragma unroll
        for (int j = 0; j < 8; ++j) Vt[(seg + j) * VTS + t] = vv[j];
    }
    wait_async0();
    __syncthreads();

    // phase 1: S = Q K^T  (M=128 rows split across 8 waves, N=128, K=64)
    {
        v8f z = {0.f, 0.f, 0.f, 0.f, 0.f, 0.f, 0.f, 0.f};
        v8f acc[8];
#pragma unroll
        for (int nt = 0; nt < 8; ++nt) acc[nt] = z;
#pragma unroll
        for (int kb = 0; kb < 64; kb += 32) {
            int m   = wave * 16 + l15;
            int klo = hi * 16;
            const _Float16* ap = &Qs[m * QS + kb + klo];
            v16h a = mk16(*(const v8h*)ap, *(const v8h*)(ap + 8));
#pragma unroll
            for (int nt = 0; nt < 8; ++nt) {
                int n = nt * 16 + l15;
                const _Float16* bp2 = &Ks[n * QS + kb + hi * 16];
                v16h bf = mk16(*(const v8h*)bp2, *(const v8h*)(bp2 + 8));
                acc[nt] = __builtin_amdgcn_wmma_f32_16x16x32_f16(
                    false, a, false, bf, (short)0, acc[nt], false, false);
            }
        }
#pragma unroll
        for (int nt = 0; nt < 8; ++nt) {
            int n = nt * 16 + l15;
#pragma unroll
            for (int v = 0; v < 8; ++v)
                Sb[(wave * 16 + hi * 8 + v) * SS + n] = acc[nt][v];
        }
    }
    __syncthreads();

    // softmax over rows (scale = 1/sqrt(64)); write P as f16 into Pb
    if (tid < TT) {
        float* srow = &Sb[tid * SS];
        const float scale = 0.125f;
        float mx = -1e30f;
        for (int i = 0; i < TT; ++i) { float v = srow[i] * scale; mx = v > mx ? v : mx; }
        float sum = 0.f;
        for (int i = 0; i < TT; ++i) { float e = expf(srow[i] * scale - mx); srow[i] = e; sum += e; }
        float inv = 1.f / sum;
        for (int i = 0; i < TT; ++i) Pb[tid * PS + i] = (_Float16)(srow[i] * inv);
    }
    __syncthreads();

    // phase 2: O = P V  (M=128, N=64, K=128)
    {
        v8f z = {0.f, 0.f, 0.f, 0.f, 0.f, 0.f, 0.f, 0.f};
        v8f oacc[4];
#pragma unroll
        for (int nt = 0; nt < 4; ++nt) oacc[nt] = z;
#pragma unroll
        for (int kb = 0; kb < 128; kb += 32) {
            int m   = wave * 16 + l15;
            int klo = hi * 16;
            const _Float16* ap = &Pb[m * PS + kb + klo];
            v16h a = mk16(*(const v8h*)ap, *(const v8h*)(ap + 8));
#pragma unroll
            for (int nt = 0; nt < 4; ++nt) {
                int n = nt * 16 + l15;
                const _Float16* bp2 = &Vt[n * VTS + kb + hi * 16];
                v16h bf = mk16(*(const v8h*)bp2, *(const v8h*)(bp2 + 8));
                oacc[nt] = __builtin_amdgcn_wmma_f32_16x16x32_f16(
                    false, a, false, bf, (short)0, oacc[nt], false, false);
            }
        }
#pragma unroll
        for (int nt = 0; nt < 4; ++nt) {
            int d = nt * 16 + l15;
#pragma unroll
            for (int v = 0; v < 8; ++v) {
                int t = wave * 16 + hi * 8 + v;
                Oh[rowbase + (size_t)t * N_EMBD + d] = (_Float16)oacc[nt][v];
            }
        }
    }
}

// ---------------------------------------------------------------------------
// Residual add + LayerNorm, wave-per-row (wave32), writes fp32 + f16 operand
// ---------------------------------------------------------------------------
__global__ __launch_bounds__(256) void k_res_ln(const float* __restrict__ Xin,
                                                const float* __restrict__ Yin,
                                                const float* __restrict__ g,
                                                const float* __restrict__ bta,
                                                float* __restrict__ Xout,
                                                _Float16* __restrict__ Xh) {
    int lane = threadIdx.x & 31;
    int wave = threadIdx.x >> 5;
    int row  = blockIdx.x * 8 + wave;
    const float* xr = Xin + (size_t)row * N_EMBD;
    const float* yr = Yin ? Yin + (size_t)row * N_EMBD : nullptr;
    float xv[16];
    float s = 0.f;
#pragma unroll
    for (int j = 0; j < 16; ++j) {
        int c = lane + j * 32;
        float v = xr[c] + (yr ? yr[c] : 0.f);
        xv[j] = v; s += v;
    }
#pragma unroll
    for (int off = 16; off > 0; off >>= 1) s += __shfl_xor(s, off, 32);
    float mean = s * (1.f / N_EMBD);
    float vs = 0.f;
#pragma unroll
    for (int j = 0; j < 16; ++j) { float d = xv[j] - mean; vs += d * d; }
#pragma unroll
    for (int off = 16; off > 0; off >>= 1) vs += __shfl_xor(vs, off, 32);
    float rstd = rsqrtf(vs * (1.f / N_EMBD) + EPSV);
#pragma unroll
    for (int j = 0; j < 16; ++j) {
        int c = lane + j * 32;
        float o = (xv[j] - mean) * rstd * g[c] + bta[c];
        Xout[(size_t)row * N_EMBD + c] = o;
        Xh[(size_t)row * N_EMBD + c]  = (_Float16)o;
    }
}

// ---------------------------------------------------------------------------
// Conv 3x3 stride 2 pad 1 (+bias), direct, NCHW fp32
// ---------------------------------------------------------------------------
__global__ __launch_bounds__(256) void k_conv3x3s2(
        const float* __restrict__ in, const float* __restrict__ w,
        const float* __restrict__ bias, float* __restrict__ out,
        int Ci, int Hi, int Wi, int Co, int Ho, int Wo) {
    int i = blockIdx.x * 256 + threadIdx.x;
    int total = BATCH * Co * Ho * Wo;
    if (i >= total) return;
    int x = i % Wo; int tmp = i / Wo;
    int y = tmp % Ho; tmp /= Ho;
    int co = tmp % Co; int n = tmp / Co;
    float acc = bias[co];
    for (int ci = 0; ci < Ci; ++ci) {
        const float* ip = in + (size_t)(n * Ci + ci) * Hi * Wi;
        const float* wp = w + (size_t)(co * Ci + ci) * 9;
#pragma unroll
        for (int ky = 0; ky < 3; ++ky) {
            int iy = y * 2 - 1 + ky;
            if ((unsigned)iy >= (unsigned)Hi) continue;
#pragma unroll
            for (int kx = 0; kx < 3; ++kx) {
                int ix = x * 2 - 1 + kx;
                if ((unsigned)ix >= (unsigned)Wi) continue;
                acc += ip[iy * Wi + ix] * wp[ky * 3 + kx];
            }
        }
    }
    out[i] = acc;
}

__global__ __launch_bounds__(256) void k_zero(float* p, int n) {
    int i = blockIdx.x * 256 + threadIdx.x;
    if (i < n) p[i] = 0.f;
}

// BN stats: sums[c]=sum, sums[C+c]=sumsq over (N,H,W)
__global__ __launch_bounds__(256) void k_bnstats(const float* __restrict__ in,
                                                 float* __restrict__ sums,
                                                 int C, int HW) {
    int c = blockIdx.y;
    int per = BATCH * HW;
    float s = 0.f, s2 = 0.f;
    for (int p = blockIdx.x * 256 + threadIdx.x; p < per; p += gridDim.x * 256) {
        int n = p / HW, r = p % HW;
        float v = in[(size_t)(n * C + c) * HW + r];
        s += v; s2 += v * v;
    }
    __shared__ float sh1[256], sh2[256];
    sh1[threadIdx.x] = s; sh2[threadIdx.x] = s2;
    __syncthreads();
    for (int off = 128; off > 0; off >>= 1) {
        if (threadIdx.x < off) {
            sh1[threadIdx.x] += sh1[threadIdx.x + off];
            sh2[threadIdx.x] += sh2[threadIdx.x + off];
        }
        __syncthreads();
    }
    if (threadIdx.x == 0) {
        atomicAdd(&sums[c], sh1[0]);
        atomicAdd(&sums[C + c], sh2[0]);
    }
}

__global__ __launch_bounds__(256) void k_bnapply_elu(float* __restrict__ data,
                                                     const float* __restrict__ sums,
                                                     const float* __restrict__ g,
                                                     const float* __restrict__ b,
                                                     int C, int HW) {
    int i = blockIdx.x * 256 + threadIdx.x;
    int total = BATCH * C * HW;
    if (i >= total) return;
    int c = (i / HW) % C;
    float cnt = (float)(BATCH * HW);
    float m   = sums[c] / cnt;
    float var = sums[C + c] / cnt - m * m;
    float v = (data[i] - m) * rsqrtf(var + EPSV) * g[c] + b[c];
    data[i] = v > 0.f ? v : expm1f(v);
}

// Small dense layers (head): out = in @ w + b
__global__ __launch_bounds__(256) void k_linear(const float* __restrict__ in,
                                                const float* __restrict__ w,
                                                const float* __restrict__ b,
                                                float* __restrict__ out,
                                                int Kd, int Nd) {
    int i = blockIdx.x * 256 + threadIdx.x;
    if (i >= BATCH * Nd) return;
    int r = i / Nd, j = i % Nd;
    float acc = b[j];
    for (int k = 0; k < Kd; ++k) acc += in[r * Kd + k] * w[k * Nd + j];
    out[i] = acc;
}

// ---------------------------------------------------------------------------
// Host orchestration
// ---------------------------------------------------------------------------
extern "C" void kernel_launch(void* const* d_in, const int* in_sizes, int n_in,
                              void* d_out, int out_size, void* d_ws, size_t ws_size,
                              hipStream_t stream) {
    (void)in_sizes; (void)n_in; (void)out_size; (void)ws_size;
    const int*   index   = (const int*)  d_in[0];
    const float* tok_emb = (const float*)d_in[1];
    const float* pos_emb = (const float*)d_in[2];
    const float* Wq  = (const float*)d_in[3];
    const float* Wk  = (const float*)d_in[4];
    const float* Wv  = (const float*)d_in[5];
    const float* Wp  = (const float*)d_in[6];
    const float* bp  = (const float*)d_in[7];
    const float* W1  = (const float*)d_in[8];
    const float* b1  = (const float*)d_in[9];
    const float* W2  = (const float*)d_in[10];
    const float* b2  = (const float*)d_in[11];
    const float* ln1_g = (const float*)d_in[12];
    const float* ln1_b = (const float*)d_in[13];
    const float* ln2_g = (const float*)d_in[14];
    const float* ln2_b = (const float*)d_in[15];
    const float* lnf_g = (const float*)d_in[16];
    const float* lnf_b = (const float*)d_in[17];
    const float* conv_w[4] = {(const float*)d_in[18], (const float*)d_in[19],
                              (const float*)d_in[20], (const float*)d_in[21]};
    const float* conv_b[4] = {(const float*)d_in[22], (const float*)d_in[23],
                              (const float*)d_in[24], (const float*)d_in[25]};
    const float* bn_g[4]   = {(const float*)d_in[26], (const float*)d_in[27],
                              (const float*)d_in[28], (const float*)d_in[29]};
    const float* bn_b[4]   = {(const float*)d_in[30], (const float*)d_in[31],
                              (const float*)d_in[32], (const float*)d_in[33]};
    const float* lin_w[4]  = {(const float*)d_in[34], (const float*)d_in[35],
                              (const float*)d_in[36], (const float*)d_in[37]};
    const float* lin_b[4]  = {(const float*)d_in[38], (const float*)d_in[39],
                              (const float*)d_in[40], (const float*)d_in[41]};
    const float* fin_w = (const float*)d_in[42];
    const float* fin_b = (const float*)d_in[43];
    float* out = (float*)d_out;

    // ---- workspace bump allocator (256B aligned) ----
    char* base = (char*)d_ws;
    size_t cur = 0;
    auto alloc = [&](size_t bytes) -> char* {
        cur = (cur + 255) & ~(size_t)255;
        char* p = base + cur;
        cur += bytes;
        return p;
    };
    const size_t WSQ  = (size_t)NLAYERS * N_EMBD * N_EMBD;   // 2M elems
    const size_t WFF  = (size_t)NLAYERS * N_EMBD * FF;       // 8M elems
    _Float16* Wqh  = (_Float16*)alloc(WSQ * 2);
    _Float16* Wkh  = (_Float16*)alloc(WSQ * 2);
    _Float16* Wvh  = (_Float16*)alloc(WSQ * 2);
    _Float16* Wph  = (_Float16*)alloc(WSQ * 2);
    _Float16* W1h  = (_Float16*)alloc(WFF * 2);
    _Float16* W2h  = (_Float16*)alloc(WFF * 2);
    float*    X    = (float*)   alloc((size_t)ROWS * N_EMBD * 4);
    float*    Y    = (float*)   alloc((size_t)ROWS * N_EMBD * 4);
    _Float16* Xh   = (_Float16*)alloc((size_t)ROWS * N_EMBD * 2);
    _Float16* Qhb  = (_Float16*)alloc((size_t)ROWS * N_EMBD * 2);
    _Float16* Khb  = (_Float16*)alloc((size_t)ROWS * N_EMBD * 2);
    _Float16* Vhb  = (_Float16*)alloc((size_t)ROWS * N_EMBD * 2);
    _Float16* Atth = (_Float16*)alloc((size_t)ROWS * N_EMBD * 2);
    _Float16* H1h  = (_Float16*)alloc((size_t)ROWS * FF * 2);
    float*    convA = (float*)  alloc((size_t)BATCH * 64 * 64 * 256 * 4);  // 1 GB
    float*    convB = (float*)  alloc((size_t)BATCH * 16 * 32 * 128 * 4);  // 64 MB
    float*    bnsums = (float*) alloc(512);
    float*    linA = (float*)   alloc((size_t)BATCH * 128 * 4);
    float*    linB = (float*)   alloc((size_t)BATCH * 32 * 4);

    // ---- weight prep (f16 pack/cast, rerun every call: deterministic) ----
    k_pack_qkv<<<(int)(WSQ / 256), 256, 0, stream>>>(Wq, Wqh);
    k_pack_qkv<<<(int)(WSQ / 256), 256, 0, stream>>>(Wk, Wkh);
    k_pack_qkv<<<(int)(WSQ / 256), 256, 0, stream>>>(Wv, Wvh);
    k_cast_f16<<<(int)(WSQ / 256), 256, 0, stream>>>(Wp, Wph, (int)WSQ);
    k_cast_f16<<<(int)(WFF / 256), 256, 0, stream>>>(W1, W1h, (int)WFF);
    k_cast_f16<<<(int)(WFF / 256), 256, 0, stream>>>(W2, W2h, (int)WFF);

    // ---- embedding ----
    k_embed<<<(ROWS * N_EMBD) / 256, 256, 0, stream>>>(index, tok_emb, pos_emb, X, Xh);

    // ---- transformer layers ----
    dim3 gN512(N_EMBD / BN, ROWS / BM);
    dim3 gN2048(FF / BN, ROWS / BM);
    for (int l = 0; l < NLAYERS; ++l) {
        const size_t oSQ = (size_t)l * N_EMBD * N_EMBD;
        const size_t oFF = (size_t)l * N_EMBD * FF;
        wmma_gemm<<<gN512, 256, 0, stream>>>(Xh, Wqh + oSQ, nullptr, nullptr, Qhb,
                                             ROWS, N_EMBD, N_EMBD, 0);
        wmma_gemm<<<gN512, 256, 0, stream>>>(Xh, Wkh + oSQ, nullptr, nullptr, Khb,
                                             ROWS, N_EMBD, N_EMBD, 0);
        wmma_gemm<<<gN512, 256, 0, stream>>>(Xh, Wvh + oSQ, nullptr, nullptr, Vhb,
                                             ROWS, N_EMBD, N_EMBD, 0);
        k_attention<<<BATCH * NHEAD, 256, ATT_LDS, stream>>>(Qhb, Khb, Vhb, Atth);
        wmma_gemm<<<gN512, 256, 0, stream>>>(Atth, Wph + oSQ, bp + l * N_EMBD,
                                             Y, nullptr, ROWS, N_EMBD, N_EMBD, 0);
        k_res_ln<<<ROWS / 8, 256, 0, stream>>>(X, Y, ln1_g + l * N_EMBD,
                                               ln1_b + l * N_EMBD, X, Xh);
        wmma_gemm<<<gN2048, 256, 0, stream>>>(Xh, W1h + oFF, b1 + l * FF,
                                              nullptr, H1h, ROWS, FF, N_EMBD, 1);
        wmma_gemm<<<gN512, 256, 0, stream>>>(H1h, W2h + oFF, b2 + l * N_EMBD,
                                             Y, nullptr, ROWS, N_EMBD, FF, 0);
        k_res_ln<<<ROWS / 8, 256, 0, stream>>>(X, Y, ln2_g + l * N_EMBD,
                                               ln2_b + l * N_EMBD, X, Xh);
    }
    k_res_ln<<<ROWS / 8, 256, 0, stream>>>(X, nullptr, lnf_g, lnf_b, X, Xh);

    // ---- conv head: X viewed as NCHW (256,1,128,512) ----
    const int CoArr[4] = {64, 16, 8, 2};
    int Ci = 1, Hi = 128, Wi = 512;
    float* cin = X;
    for (int s = 0; s < 4; ++s) {
        int Co = CoArr[s];
        int Ho = (Hi - 1) / 2 + 1;
        int Wo = (Wi - 1) / 2 + 1;
        float* cout = (s % 2 == 0) ? convA : convB;
        int total = BATCH * Co * Ho * Wo;
        k_conv3x3s2<<<(total + 255) / 256, 256, 0, stream>>>(cin, conv_w[s], conv_b[s],
                                                             cout, Ci, Hi, Wi, Co, Ho, Wo);
        k_zero<<<1, 256, 0, stream>>>(bnsums, 2 * Co);
        int HW = Ho * Wo;
        dim3 gs(256, Co);
        k_bnstats<<<gs, 256, 0, stream>>>(cout, bnsums, Co, HW);
        k_bnapply_elu<<<(total + 255) / 256, 256, 0, stream>>>(cout, bnsums,
                                                               bn_g[s], bn_b[s], Co, HW);
        cin = cout; Ci = Co; Hi = Ho; Wi = Wo;
    }
    // final conv activations in convB: (256, 2*8*32=512) row-major flat

    // ---- linear head ----
    k_linear<<<(BATCH * 128 + 255) / 256, 256, 0, stream>>>(convB, lin_w[0], lin_b[0],
                                                            linA, 512, 128);
    k_linear<<<(BATCH * 32 + 255) / 256, 256, 0, stream>>>(linA, lin_w[1], lin_b[1],
                                                           linB, 128, 32);
    k_linear<<<(BATCH * 16 + 255) / 256, 256, 0, stream>>>(linB, lin_w[2], lin_b[2],
                                                           linA, 32, 16);
    k_linear<<<(BATCH * 8 + 255) / 256, 256, 0, stream>>>(linA, lin_w[3], lin_b[3],
                                                          linB, 16, 8);
    k_linear<<<(BATCH * 3 + 255) / 256, 256, 0, stream>>>(linB, fin_w, fin_b,
                                                          out, 8, 3);
}
